// TSSAttention_89704686944456
// MI455X (gfx1250) — matrix operand order repaired and checked by reading, not compiled, not run on hardware
//
#include <hip/hip_runtime.h>

typedef __attribute__((ext_vector_type(16))) __bf16 v16bf;
typedef __attribute__((ext_vector_type(8)))  float  v8f;
typedef __attribute__((ext_vector_type(4)))  int    v4i;

#define BATCH  4
#define NSEQ   8192
#define DMODEL 768
#define HN     12
#define DHN    64
#define MROWS  (BATCH * NSEQ)   // 32768

#define AS1 __attribute__((address_space(1)))
#define AS3 __attribute__((address_space(3)))

// gfx1250 async global->LDS path (ASYNCcnt-tracked). Guarded so the file
// compiles on toolchains that lack the builtins (falls back to sync copy).
#if defined(__has_builtin)
#if __has_builtin(__builtin_amdgcn_global_load_async_to_lds_b128) && \
    __has_builtin(__builtin_amdgcn_s_wait_asynccnt)
#define USE_ASYNC 1
#else
#define USE_ASYNC 0
#endif
#else
#define USE_ASYNC 0
#endif

__device__ __forceinline__ unsigned short f2bf(float f) {
  unsigned u = __builtin_bit_cast(unsigned, f);
  u += 0x7FFFu + ((u >> 16) & 1u);           // round-to-nearest-even
  return (unsigned short)(u >> 16);
}

// ---------------- f32 -> bf16 cast ----------------
__global__ void k_f32_to_bf16(const float* __restrict__ src,
                              unsigned short* __restrict__ dst, int n) {
  for (int i = blockIdx.x * blockDim.x + threadIdx.x; i < n;
       i += gridDim.x * blockDim.x)
    dst[i] = f2bf(src[i]);
}

// ------- f32 [K][N] -> bf16 transposed [N][K] (weights; one-time, tiny) ----
__global__ void k_f32_to_bf16_T(const float* __restrict__ src,
                                unsigned short* __restrict__ dst,
                                int Nn, int K) {
  const int i = blockIdx.x * blockDim.x + threadIdx.x;   // over N*K
  if (i >= Nn * K) return;
  const int n = i / K;
  const int k = i - n * K;
  dst[i] = f2bf(src[(size_t)k * Nn + n]);
}

// 32-byte copy of one 16-bf16 segment into LDS (async when available)
__device__ __forceinline__ void seg_copy(const unsigned short* g,
                                         unsigned short* l) {
#if USE_ASYNC
  __builtin_amdgcn_global_load_async_to_lds_b128((AS1 v4i*)g, (AS3 v4i*)l, 0, 0);
  __builtin_amdgcn_global_load_async_to_lds_b128((AS1 v4i*)(g + 8),
                                                 (AS3 v4i*)(l + 8), 0, 0);
#else
  const uint4* s = (const uint4*)g;
  uint4* d = (uint4*)l;
  d[0] = s[0];
  d[1] = s[1];
#endif
}

// ---------------- bf16 WMMA GEMM: C[M,N] = A[M,K] * Bt[N,K]^T (+bias) -------
// A row-major [M][K], Bt row-major [N][K] (pre-transposed weights).
// Block tile 128x128, K-step 32, double-buffered LDS. 256 threads = 8 waves.
// Waves: wq = waveId&3 -> 32-row M quadrant, nq = waveId>>2 -> 64-col N half.
// Each wave: 2 (M) x 4 (N) tiles of 16x16 -> 8 v8f accumulators.
__global__ __launch_bounds__(256)
void k_gemm_bf16(const unsigned short* __restrict__ A,
                 const unsigned short* __restrict__ Bt,
                 float* __restrict__ C,
                 const float* __restrict__ bias,
                 int M, int Nn, int K) {
  __shared__ __align__(16) unsigned short ldsA[2][128 * 32];  // [buf][row][k]
  __shared__ __align__(16) unsigned short ldsB[2][128 * 32];  // [buf][col][k]

  const int tid    = threadIdx.x;
  const int lane   = tid & 31;
  const int waveId = tid >> 5;
  const int half   = lane >> 4;    // lane group 0/1
  const int mrow   = lane & 15;
  const int wq     = waveId & 3;
  const int nq     = waveId >> 2;
  const int blockM = blockIdx.y * 128;
  const int blockN = blockIdx.x * 128;

  v8f acc[2][4];
#pragma unroll
  for (int i = 0; i < 2; ++i)
#pragma unroll
    for (int j = 0; j < 4; ++j)
#pragma unroll
      for (int r = 0; r < 8; ++r) acc[i][j][r] = 0.0f;

  // tile loader role: each thread fills one 16-element segment of A and B
  const int row = tid >> 1;        // 0..127 (M row of A tile / N col of B tile)
  const int seg = tid & 1;         // K segment (0 or 1) * 16 elements
  const unsigned short* gA = A + (size_t)(blockM + row) * K + seg * 16;
  const unsigned short* gB = Bt + (size_t)(blockN + row) * K + seg * 16;
  const int ldsOff = row * 32 + seg * 16;

  // prologue: fill buffer 0
  seg_copy(gA, &ldsA[0][ldsOff]);
  seg_copy(gB, &ldsB[0][ldsOff]);

  for (int kt = 0; kt < K; kt += 32) {
    const int cur = (kt >> 5) & 1;
#if USE_ASYNC
    __builtin_amdgcn_s_wait_asynccnt(0);
#endif
    __syncthreads();
    if (kt + 32 < K) {             // overlap next tile fill with WMMAs
      const int nxt = cur ^ 1;
      seg_copy(gA + kt + 32, &ldsA[nxt][ldsOff]);
      seg_copy(gB + kt + 32, &ldsB[nxt][ldsOff]);
    }

    // A fragments: contiguous K order matching B:
    //   VGPR v : K = half*16 + 2v,2v+1
    union VA { v16bf v; unsigned u[8]; } a[2];
#pragma unroll
    for (int mt = 0; mt < 2; ++mt) {
      const unsigned short* ap = &ldsA[cur][(wq * 32 + mt * 16 + mrow) * 32];
#pragma unroll
      for (int v = 0; v < 8; ++v) {
        const int kl = half * 16 + 2 * v;
        a[mt].u[v] = *(const unsigned*)(ap + kl);
      }
    }
    // B fragments: lanes 0-15 hold K=0..15, lanes 16-31 K=16..31, n = lane%16
#pragma unroll
    for (int nt = 0; nt < 4; ++nt) {
      union VB { v16bf v; uint4 q[2]; } bf;
      const unsigned short* bp =
          &ldsB[cur][(nq * 64 + nt * 16 + mrow) * 32 + half * 16];
      bf.q[0] = *(const uint4*)(bp);
      bf.q[1] = *(const uint4*)(bp + 8);
#pragma unroll
      for (int mt = 0; mt < 2; ++mt) {
        acc[mt][nt] = __builtin_amdgcn_wmma_f32_16x16x32_bf16(
            false, a[mt].v, false, bf.v, (short)0, acc[mt][nt], false, false);
      }
    }
  }

  // Epilogue: C/D layout — VGPR r: lanes 0-15 row M=r, lanes 16-31 row M=8+r
#pragma unroll
  for (int mt = 0; mt < 2; ++mt) {
#pragma unroll
    for (int nt = 0; nt < 4; ++nt) {
      const int gcol = blockN + nq * 64 + nt * 16 + mrow;
      const float bv = bias ? bias[gcol] : 0.0f;
#pragma unroll
      for (int r = 0; r < 8; ++r) {
        const int grow = blockM + wq * 32 + mt * 16 + half * 8 + r;
        C[(size_t)grow * Nn + gcol] = acc[mt][nt][r] + bv;
      }
    }
  }
}

// ---------------- column norms over sequence dim ----------------
// invd2[b*768+c] = 1 / max(sqrt(sum_n W[b,n,c]^2), 1e-12)^2
__global__ void k_colnorm(const float* __restrict__ W, float* __restrict__ invd2) {
  const int g = blockIdx.x * blockDim.x + threadIdx.x;   // 0..B*768-1
  const int b = g / DMODEL;
  const int c = g - b * DMODEL;
  const float* p = W + (size_t)b * NSEQ * DMODEL + c;
  float s = 0.0f;
  for (int n = 0; n < NSEQ; ++n) {
    const float w = p[(size_t)n * DMODEL];
    s += w * w;
  }
  const float d = fmaxf(sqrtf(s), 1e-12f);
  invd2[g] = 1.0f / (d * d);
}

// ---------------- per-(b,h) softmax over sequence dim ----------------
__global__ __launch_bounds__(256)
void k_softmax(const float* __restrict__ W, const float* __restrict__ invd2,
               const float* __restrict__ temp, float* __restrict__ Pi) {
  __shared__ float sl[NSEQ];       // logits / exp values (32 KB of 320 KB WGP LDS)
  __shared__ float sred[256];
  __shared__ float sinv[DHN];
  const int bh = blockIdx.x;       // 0..B*H-1
  const int b  = bh / HN;
  const int h  = bh - b * HN;
  const int tid = threadIdx.x;
  if (tid < DHN) sinv[tid] = invd2[b * DMODEL + h * DHN + tid];
  __syncthreads();
  const float th = temp[h];

  float lmax = -1e30f;
  for (int n = tid; n < NSEQ; n += 256) {
    const float4* wp = (const float4*)(W + ((size_t)(b * NSEQ + n)) * DMODEL + h * DHN);
    float s = 0.0f;
#pragma unroll
    for (int q = 0; q < 16; ++q) {
      const float4 v = wp[q];
      s += v.x * v.x * sinv[4 * q + 0] + v.y * v.y * sinv[4 * q + 1] +
           v.z * v.z * sinv[4 * q + 2] + v.w * v.w * sinv[4 * q + 3];
    }
    const float lg = th * s;
    sl[n] = lg;
    lmax = fmaxf(lmax, lg);
  }
  sred[tid] = lmax;
  __syncthreads();
  for (int o = 128; o > 0; o >>= 1) {
    if (tid < o) sred[tid] = fmaxf(sred[tid], sred[tid + o]);
    __syncthreads();
  }
  const float gmax = sred[0];
  __syncthreads();

  float lsum = 0.0f;
  for (int n = tid; n < NSEQ; n += 256) {
    const float e = __expf(sl[n] - gmax);
    sl[n] = e;
    lsum += e;
  }
  sred[tid] = lsum;
  __syncthreads();
  for (int o = 128; o > 0; o >>= 1) {
    if (tid < o) sred[tid] += sred[tid + o];
    __syncthreads();
  }
  const float inv = 1.0f / sred[0];
  float* po = Pi + (size_t)(b * HN + h) * NSEQ;
  for (int n = tid; n < NSEQ; n += 256) po[n] = sl[n] * inv;
}

// ---------------- dots -> attn = 1/(1+dots) ----------------
__global__ void k_attn(const float* __restrict__ W, const float* __restrict__ Pi,
                       float* __restrict__ attn) {
  const int g = blockIdx.x * blockDim.x + threadIdx.x;   // 0..B*768-1
  const int b = g / DMODEL;
  const int c = g - b * DMODEL;
  const int h = c >> 6;
  const float* p  = W + (size_t)b * NSEQ * DMODEL + c;
  const float* pp = Pi + (size_t)(b * HN + h) * NSEQ;
  float s = 0.0f;
  for (int n = 0; n < NSEQ; ++n) {
    const float w = p[(size_t)n * DMODEL];
    s += pp[n] * w * w;
  }
  const float d = s / (1.0f + 1e-8f);   // Pi_norm = Pi / (sum Pi + 1e-8), sum==1
  attn[g] = 1.0f / (1.0f + d);
}

// ---------------- y = -w * Pi * attn (bf16 out for GEMM2) ----------------
// one block per (b, n) row; 256 threads x 3 elements = 768 columns
__global__ __launch_bounds__(256)
void k_make_y(const float* __restrict__ W, const float* __restrict__ Pi,
              const float* __restrict__ attn, unsigned short* __restrict__ Y) {
  const int rowg = blockIdx.x;            // 0..MROWS-1
  const int b = rowg / NSEQ;
  const int n = rowg - b * NSEQ;
  const int tid = threadIdx.x;
  const float* wr = W + (size_t)rowg * DMODEL;
  unsigned short* yr = Y + (size_t)rowg * DMODEL;
  const float* at = attn + b * DMODEL;
  const float* pb = Pi + (size_t)b * HN * NSEQ + n;     // stride NSEQ per head
#pragma unroll
  for (int j = 0; j < 3; ++j) {
    const int c = tid + j * 256;
    const int h = c >> 6;
    const float y = -wr[c] * pb[(size_t)h * NSEQ] * at[c];
    yr[c] = f2bf(y);
  }
}

extern "C" void kernel_launch(void* const* d_in, const int* in_sizes, int n_in,
                              void* d_out, int out_size, void* d_ws, size_t ws_size,
                              hipStream_t stream) {
  const float* x     = (const float*)d_in[0];
  const float* w_qkv = (const float*)d_in[1];
  const float* temp  = (const float*)d_in[2];
  const float* w_out = (const float*)d_in[3];
  const float* b_out = (const float*)d_in[4];
  float* out = (float*)d_out;

  char* ws = (char*)d_ws;
  size_t off = 0;
  auto alloc = [&](size_t bytes) -> void* {
    void* p = ws + off;
    off = (off + bytes + 255) & ~(size_t)255;
    return p;
  };

  float* Wbuf          = (float*)alloc((size_t)MROWS * DMODEL * 4);          // 100 MB
  unsigned short* xbf  = (unsigned short*)alloc((size_t)MROWS * DMODEL * 2); // 50 MB (reused as Y)
  unsigned short* qbfT = (unsigned short*)alloc((size_t)DMODEL * DMODEL * 2);
  unsigned short* obfT = (unsigned short*)alloc((size_t)DMODEL * DMODEL * 2);
  float* invd2         = (float*)alloc((size_t)BATCH * DMODEL * 4);
  float* piBuf         = (float*)alloc((size_t)BATCH * HN * NSEQ * 4);
  float* attnBuf       = (float*)alloc((size_t)BATCH * DMODEL * 4);

  // 1) casts (weights transposed to [N][K] so GEMM B-tiles are K-major rows)
  k_f32_to_bf16<<<4096, 256, 0, stream>>>(x, xbf, MROWS * DMODEL);
  k_f32_to_bf16_T<<<(DMODEL * DMODEL + 255) / 256, 256, 0, stream>>>(
      w_qkv, qbfT, DMODEL, DMODEL);
  k_f32_to_bf16_T<<<(DMODEL * DMODEL + 255) / 256, 256, 0, stream>>>(
      w_out, obfT, DMODEL, DMODEL);

  // 2) W = x @ w_qkv   (f32 accum, bf16 WMMA)
  k_gemm_bf16<<<dim3(DMODEL / 128, MROWS / 128), 256, 0, stream>>>(
      xbf, qbfT, Wbuf, nullptr, MROWS, DMODEL, DMODEL);

  // 3) column norms over sequence dim
  k_colnorm<<<(BATCH * DMODEL) / 256, 256, 0, stream>>>(Wbuf, invd2);

  // 4) per-(b,h) softmax over sequence
  k_softmax<<<BATCH * HN, 256, 0, stream>>>(Wbuf, invd2, temp, piBuf);

  // 5) dots / attn
  k_attn<<<(BATCH * DMODEL) / 256, 256, 0, stream>>>(Wbuf, piBuf, attnBuf);

  // 6) y = -w * Pi * attn  (bf16, reuses xbf region)
  k_make_y<<<MROWS, 256, 0, stream>>>(Wbuf, piBuf, attnBuf, xbf);

  // 7) out = y @ w_out + b_out
  k_gemm_bf16<<<dim3(DMODEL / 128, MROWS / 128), 256, 0, stream>>>(
      xbf, obfT, out, b_out, MROWS, DMODEL, DMODEL);
}
